// MultiScaleRetention_10376640987876
// MI455X (gfx1250) — matrix operand order repaired and checked by reading, not compiled, not run on hardware
//
#include <hip/hip_runtime.h>

// ---------------- problem constants ----------------
constexpr int B_ = 2;
constexpr int N_ = 2048;
constexpr int E_ = 512;
constexpr int H_ = 8;
constexpr int D_ = 64;          // E/H
constexpr int M_ = B_ * N_;     // 4096 rows for all projections

// ---------------- vector types ----------------
typedef __bf16 bf16_t;
typedef __attribute__((ext_vector_type(8)))  __bf16 v8bf;
typedef __attribute__((ext_vector_type(16))) __bf16 v16bf;
typedef __attribute__((ext_vector_type(8)))  float  v8f;
typedef __attribute__((ext_vector_type(4)))  unsigned int u32x4;
typedef __attribute__((ext_vector_type(8)))  int i32x8;
typedef __attribute__((ext_vector_type(4)))  int i32x4;

// float -> bf16, round-to-nearest-even
static __device__ __forceinline__ bf16_t f2bf(float f) {
  unsigned u = __builtin_bit_cast(unsigned, f);
  u += 0x7FFFu + ((u >> 16) & 1u);
  unsigned short h = (unsigned short)(u >> 16);
  return __builtin_bit_cast(bf16_t, h);
}

// assemble one 16-element bf16 fragment from two contiguous 8-element chunks
static __device__ __forceinline__ v16bf ldfrag(const bf16_t* p0, const bf16_t* p1) {
  v8bf lo = *(const v8bf*)p0;
  v8bf hi = *(const v8bf*)p1;
  v16bf r;
#pragma unroll
  for (int i = 0; i < 8; ++i) { r[i] = lo[i]; r[i + 8] = hi[i]; }
  return r;
}

static __device__ __forceinline__ v8f wmma_bf16(v16bf a, v16bf b, v8f c) {
  return __builtin_amdgcn_wmma_f32_16x16x32_bf16(false, a, false, b, (short)0, c, false, false);
}

// ---------------- elementwise kernels ----------------
// vectorized fp32 -> bf16 (8 elements / thread, n must be a multiple of 8)
__global__ void f2bf_kernel(const float* __restrict__ src, bf16_t* __restrict__ dst, int n) {
  int i = (blockIdx.x * blockDim.x + threadIdx.x) * 8;
  if (i >= n) return;
  float4 a = *(const float4*)(src + i);
  float4 b = *(const float4*)(src + i + 4);
  v8bf o;
  o[0] = f2bf(a.x); o[1] = f2bf(a.y); o[2] = f2bf(a.z); o[3] = f2bf(a.w);
  o[4] = f2bf(b.x); o[5] = f2bf(b.y); o[6] = f2bf(b.z); o[7] = f2bf(b.w);
  *(v8bf*)(dst + i) = o;
}

// xPos tables: sin/cos/scale, N_ x (D_/2)
__global__ void tables_kernel(float* __restrict__ sint, float* __restrict__ cost,
                              float* __restrict__ scalet) {
  int idx = blockIdx.x * blockDim.x + threadIdx.x;
  if (idx >= N_ * 32) return;
  int j = idx & 31;
  int n = idx >> 5;
  float xscale  = (2.0f * (float)j + 0.4f * 64.0f) / (1.4f * 64.0f);
  float power   = ((float)n - 1024.0f) * (1.0f / 512.0f);      // (n - N/2) / SCALE_BASE
  float sc      = __expf(power * __logf(xscale));
  float invfreq = __expf(-(float)j * (1.0f / 32.0f) * 9.210340372f); // 10000^(-j/half)
  float ang     = (float)n * invfreq;
  sint[idx]   = __sinf(ang);
  cost[idx]   = __cosf(ang);
  scalet[idx] = sc;
}

// rotary on q/k (+ 1/sqrt(D) on k), layout (B,N,H,D)->(B,H,N,D) bf16, V->(B,H,D,N) bf16,
// SiLU gate -> fp32 (B,N,E). One thread per (b,n,h,pair).
__global__ void xpos_pack_kernel(const float* __restrict__ qraw, const float* __restrict__ kraw,
                                 const float* __restrict__ vraw, const float* __restrict__ graw,
                                 const float* __restrict__ sint, const float* __restrict__ cost,
                                 const float* __restrict__ scalet,
                                 bf16_t* __restrict__ qrot, bf16_t* __restrict__ krot,
                                 bf16_t* __restrict__ vT,   float* __restrict__ gate) {
  int idx = blockIdx.x * blockDim.x + threadIdx.x;
  if (idx >= B_ * N_ * H_ * 32) return;
  int j = idx & 31;
  int h = (idx >> 5) & (H_ - 1);
  int n = (idx >> 8) & (N_ - 1);
  int b = idx >> 19;

  float sn = sint[n * 32 + j], cs = cost[n * 32 + j], sc = scalet[n * 32 + j];
  size_t src = ((size_t)(b * N_ + n)) * E_ + h * D_ + 2 * j;
  size_t dst = ((size_t)((b * H_ + h) * N_) + n) * D_ + 2 * j;

  // q with scale
  float x1 = qraw[src], x2 = qraw[src + 1];
  float sq = sn * sc, cq = cs * sc;
  qrot[dst]     = f2bf(x1 * cq - x2 * sq);
  qrot[dst + 1] = f2bf(x2 * cq + x1 * sq);

  // k with 1/scale, then / sqrt(D)
  float isc = 1.0f / sc;
  float sk = sn * isc, ck = cs * isc;
  float k1 = kraw[src], k2 = kraw[src + 1];
  krot[dst]     = f2bf((k1 * ck - k2 * sk) * 0.125f);
  krot[dst + 1] = f2bf((k2 * ck + k1 * sk) * 0.125f);

  // v transposed per head: (D, N)
  size_t vdst = ((size_t)((b * H_ + h) * D_) + 2 * j) * N_ + n;
  vT[vdst]      = f2bf(vraw[src]);
  vT[vdst + N_] = f2bf(vraw[src + 1]);

  // silu gate (fp32)
  float g1 = graw[src], g2 = graw[src + 1];
  gate[src]     = g1 / (1.0f + __expf(-g1));
  gate[src + 1] = g2 / (1.0f + __expf(-g2));
}

// ---------------- generic WMMA GEMM: C(MxE) = A(MxE) @ W(ExE)^T + bias ----------------
// grid.x = M/16 ; blockDim = 256 (8 waves); wave w covers cols [w*64, w*64+64).
// The block's 16xE A tile is DMA'd to LDS once by the Tensor Data Mover; W streams
// from global (L2-resident: reused by all 256 row-blocks).
__global__ void gemm_bf16_kernel(const bf16_t* __restrict__ A, const bf16_t* __restrict__ W,
                                 const float* __restrict__ bias, float* __restrict__ C) {
  __shared__ bf16_t ldsA[16 * E_];   // 16 KB

  const int lane = threadIdx.x & 31;
  const int wave = threadIdx.x >> 5;
  const int l16  = lane & 15;
  const int half = lane >> 4;
  const int m0   = blockIdx.x * 16;
  const int n0   = wave * 64;

  // ---- TDM: stage the 16 x 512 bf16 A tile into LDS (issued by wave 0 only) ----
  if (threadIdx.x < 32) {
    unsigned lds_off = (unsigned)(uintptr_t)(&ldsA[0]);       // LDS aperture: low 32 bits = offset
    unsigned long long ga = (unsigned long long)(uintptr_t)(A + (size_t)m0 * E_);
    u32x4 g0;
    g0[0] = 1u;                                               // count=1 (valid user descriptor)
    g0[1] = lds_off;                                          // lds_addr
    g0[2] = (unsigned)(ga & 0xFFFFFFFFu);                     // global_addr[31:0]
    g0[3] = (unsigned)((ga >> 32) & 0x01FFFFFFu) | (2u << 30);// global_addr[56:32] | type=2
    i32x8 g1;
    g1[0] = 1 << 16;                                          // data_size=1 (2 bytes/elem)
    g1[1] = (E_ & 0xFFFF) << 16;                              // tensor_dim0[15:0] = 512
    g1[2] = ((E_ >> 16) & 0xFFFF) | (16 << 16);               // tensor_dim0 hi | tensor_dim1 lo = 16
    g1[3] = (E_ & 0xFFFF) << 16;                              // tensor_dim1 hi=0 | tile_dim0 = 512
    g1[4] = 16;                                               // tile_dim1 = 16 | tile_dim2 = 0
    g1[5] = E_;                                               // tensor_dim0_stride lo = 512
    g1[6] = 0;                                                // stride hi | tensor_dim1_stride lo
    g1[7] = 0;                                                // tensor_dim1_stride hi
    i32x4 gz4 = {0, 0, 0, 0};                                 // groups 2/3 unused (2-D tensor)
    i32x8 gz8 = {0, 0, 0, 0, 0, 0, 0, 0};
    __builtin_amdgcn_tensor_load_to_lds(g0, g1, gz4, gz4, gz8, 0);
    __builtin_amdgcn_s_wait_tensorcnt((short)0);
  }
  __syncthreads();

  v8f acc[4] = {};
  const bf16_t* arow = &ldsA[(size_t)l16 * E_];

  for (int k0 = 0; k0 < E_; k0 += 32) {
    // A fragment from LDS (ds_load_b128 pairs)
    v16bf a = ldfrag(arow + k0 + half * 8, arow + k0 + half * 8 + 16);
    // batch-load all four B fragments before the WMMA group (overlap loads w/ math)
    v16bf bfr[4];
#pragma unroll
    for (int t = 0; t < 4; ++t) {
      const bf16_t* wrow = W + (size_t)(n0 + t * 16 + l16) * E_ + k0 + half * 8;
      if (k0 + 32 < E_) __builtin_prefetch(wrow + 32, 0, 3);
      bfr[t] = ldfrag(wrow, wrow + 16);
    }
#pragma unroll
    for (int t = 0; t < 4; ++t) acc[t] = wmma_bf16(a, bfr[t], acc[t]);
  }
#pragma unroll
  for (int t = 0; t < 4; ++t) {
#pragma unroll
    for (int r = 0; r < 8; ++r) {
      int row = m0 + r + 8 * half;
      int col = n0 + t * 16 + l16;
      C[(size_t)row * E_ + col] = acc[t][r] + bias[col];
    }
  }
}

// ---------------- retention core ----------------
// grid = (N/16/8, B*H), block = 256. Wave w: 16 q-rows; causal loop over 32-wide s-chunks.
__global__ void retention_kernel(const bf16_t* __restrict__ qrot, const bf16_t* __restrict__ krot,
                                 const bf16_t* __restrict__ vT,  const float* __restrict__ gate,
                                 bf16_t* __restrict__ retg) {
  __shared__ bf16_t simlds[8][16 * 32];   // per-wave 16x32 bf16 staging tile

  const int lane = threadIdx.x & 31;
  const int wave = threadIdx.x >> 5;
  const int l16  = lane & 15;
  const int half = lane >> 4;
  const int bh = blockIdx.y;
  const int b  = bh >> 3;
  const int h  = bh & (H_ - 1);
  const int q0 = (blockIdx.x * 8 + wave) * 16;

  // gamma_h = 1 - exp(linspace(log(1/32), log(1/512), H))
  float t01   = (float)h * (1.0f / 7.0f);
  float lg    = -3.4657359f + t01 * (-6.2383246f + 3.4657359f);
  float gamma = 1.0f - __expf(lg);
  float logg  = __logf(gamma);

  const size_t base = (size_t)bh * N_ * D_;
  const bf16_t* qh = qrot + base;
  const bf16_t* kh = krot + base;
  const bf16_t* vh = vT + base;   // (D, N)

  // A fragments of the q tile (16 x 64 = two K=32 fragments), loaded once
  const bf16_t* qrow = qh + (size_t)(q0 + l16) * D_;
  v16bf aq0 = ldfrag(qrow + half * 8,      qrow + half * 8 + 16);
  v16bf aq1 = ldfrag(qrow + 32 + half * 8, qrow + 32 + half * 8 + 16);

  v8f acc[4] = {};

  for (int s0 = 0; s0 <= q0; s0 += 32) {
    // two 16x16 sim sub-tiles -> masked -> bf16 in LDS (row-major 16x32)
#pragma unroll
    for (int sub = 0; sub < 2; ++sub) {
      v8f sim = {};
      const bf16_t* krow = kh + (size_t)(s0 + sub * 16 + l16) * D_ + half * 8;
      v16bf bk0 = ldfrag(krow,      krow + 16);
      v16bf bk1 = ldfrag(krow + 32, krow + 48);
      sim = wmma_bf16(aq0, bk0, sim);
      sim = wmma_bf16(aq1, bk1, sim);
      int s = s0 + sub * 16 + l16;
#pragma unroll
      for (int r = 0; r < 8; ++r) {
        int n    = q0 + r + 8 * half;
        int dist = n - s;
        float m  = (dist >= 0) ? __expf(logg * (float)dist) : 0.0f;
        simlds[wave][(r + 8 * half) * 32 + sub * 16 + l16] = f2bf(sim[r] * m);
      }
    }
    // reload masked sim as an A fragment (in-wave LDS relayout; compiler waits dscnt)
    const bf16_t* srow = &simlds[wave][(size_t)l16 * 32];
    v16bf as = ldfrag(srow + half * 8, srow + half * 8 + 16);
    // ret += sim @ v  (v as (D,N) -> per-lane contiguous B fragments)
#pragma unroll
    for (int t = 0; t < 4; ++t) {
      const bf16_t* vrow = vh + (size_t)(t * 16 + l16) * N_ + s0 + half * 8;
      v16bf bv = ldfrag(vrow, vrow + 16);
      acc[t] = wmma_bf16(as, bv, acc[t]);
    }
  }

  // GroupNorm over D=64 per row + gate + bf16 store in (B,N,E) layout
#pragma unroll
  for (int r = 0; r < 8; ++r) {
    float a0 = acc[0][r], a1 = acc[1][r], a2 = acc[2][r], a3 = acc[3][r];
    float s1 = a0 + a1 + a2 + a3;
    float s2 = a0 * a0 + a1 * a1 + a2 * a2 + a3 * a3;
#pragma unroll
    for (int m = 1; m < 16; m <<= 1) {   // reduce across the 16-lane half (wave32)
      s1 += __shfl_xor(s1, m, 32);
      s2 += __shfl_xor(s2, m, 32);
    }
    float mean = s1 * (1.0f / 64.0f);
    float var  = s2 * (1.0f / 64.0f) - mean * mean;
    float rstd = rsqrtf(var + 1e-6f);
    int n = q0 + r + 8 * half;
#pragma unroll
    for (int t = 0; t < 4; ++t) {
      float x = acc[t][r];
      int e = h * D_ + t * 16 + l16;
      size_t idx = ((size_t)(b * N_ + n)) * E_ + e;
      retg[idx] = f2bf((x - mean) * rstd * gate[idx]);
    }
  }
}

// ---------------- host orchestration ----------------
extern "C" void kernel_launch(void* const* d_in, const int* in_sizes, int n_in,
                              void* d_out, int out_size, void* d_ws, size_t ws_size,
                              hipStream_t stream) {
  (void)in_sizes; (void)n_in; (void)out_size; (void)ws_size;
  const float* query = (const float*)d_in[0];
  const float* keyi  = (const float*)d_in[1];
  const float* value = (const float*)d_in[2];
  const float* Wq = (const float*)d_in[3];  const float* bq = (const float*)d_in[4];
  const float* Wk = (const float*)d_in[5];  const float* bk = (const float*)d_in[6];
  const float* Wv = (const float*)d_in[7];  const float* bv = (const float*)d_in[8];
  const float* Wg = (const float*)d_in[9];  const float* bg = (const float*)d_in[10];
  const float* Wo = (const float*)d_in[11]; const float* bo = (const float*)d_in[12];
  float* out = (float*)d_out;

  char* ws = (char*)d_ws;
  size_t off = 0;
  auto alloc = [&](size_t bytes) -> void* {
    void* p = ws + off;
    off = (off + bytes + 255) & ~(size_t)255;
    return p;
  };
  const size_t ME = (size_t)M_ * E_;   // 2,097,152 elements
  bf16_t* qbf   = (bf16_t*)alloc(ME * 2);
  bf16_t* kbf   = (bf16_t*)alloc(ME * 2);
  bf16_t* vbf   = (bf16_t*)alloc(ME * 2);
  bf16_t* Wqb   = (bf16_t*)alloc((size_t)E_ * E_ * 2);
  bf16_t* Wkb   = (bf16_t*)alloc((size_t)E_ * E_ * 2);
  bf16_t* Wvb   = (bf16_t*)alloc((size_t)E_ * E_ * 2);
  bf16_t* Wgb   = (bf16_t*)alloc((size_t)E_ * E_ * 2);
  bf16_t* Wob   = (bf16_t*)alloc((size_t)E_ * E_ * 2);
  float*  qraw  = (float*)alloc(ME * 4);
  float*  kraw  = (float*)alloc(ME * 4);
  float*  vraw  = (float*)alloc(ME * 4);
  float*  graw  = (float*)alloc(ME * 4);
  float*  sint  = (float*)alloc((size_t)N_ * 32 * 4);
  float*  cost  = (float*)alloc((size_t)N_ * 32 * 4);
  float*  sclt  = (float*)alloc((size_t)N_ * 32 * 4);
  bf16_t* qrot  = (bf16_t*)alloc(ME * 2);
  bf16_t* krot  = (bf16_t*)alloc(ME * 2);
  bf16_t* vT    = (bf16_t*)alloc(ME * 2);
  float*  gate  = (float*)alloc(ME * 4);
  bf16_t* retg  = (bf16_t*)alloc(ME * 2);

  const int CT = 256;
  const int nME = (int)ME;
  const int nEE = E_ * E_;

  // 1) fp32 -> bf16 conversions (vectorized x8)
  f2bf_kernel<<<(nME / 8 + CT - 1) / CT, CT, 0, stream>>>(query, qbf, nME);
  f2bf_kernel<<<(nME / 8 + CT - 1) / CT, CT, 0, stream>>>(keyi,  kbf, nME);
  f2bf_kernel<<<(nME / 8 + CT - 1) / CT, CT, 0, stream>>>(value, vbf, nME);
  f2bf_kernel<<<(nEE / 8 + CT - 1) / CT, CT, 0, stream>>>(Wq, Wqb, nEE);
  f2bf_kernel<<<(nEE / 8 + CT - 1) / CT, CT, 0, stream>>>(Wk, Wkb, nEE);
  f2bf_kernel<<<(nEE / 8 + CT - 1) / CT, CT, 0, stream>>>(Wv, Wvb, nEE);
  f2bf_kernel<<<(nEE / 8 + CT - 1) / CT, CT, 0, stream>>>(Wg, Wgb, nEE);
  f2bf_kernel<<<(nEE / 8 + CT - 1) / CT, CT, 0, stream>>>(Wo, Wob, nEE);

  // 2) xPos tables
  tables_kernel<<<(N_ * 32 + CT - 1) / CT, CT, 0, stream>>>(sint, cost, sclt);

  // 3) input projections (WMMA GEMMs, A tile via TDM)
  gemm_bf16_kernel<<<M_ / 16, 256, 0, stream>>>(qbf, Wqb, bq, qraw);
  gemm_bf16_kernel<<<M_ / 16, 256, 0, stream>>>(kbf, Wkb, bk, kraw);
  gemm_bf16_kernel<<<M_ / 16, 256, 0, stream>>>(vbf, Wvb, bv, vraw);
  gemm_bf16_kernel<<<M_ / 16, 256, 0, stream>>>(qbf, Wgb, bg, graw);

  // 4) rotary + pack + silu
  {
    int n = B_ * N_ * H_ * 32;
    xpos_pack_kernel<<<(n + CT - 1) / CT, CT, 0, stream>>>(
        qraw, kraw, vraw, graw, sint, cost, sclt, qrot, krot, vT, gate);
  }

  // 5) retention + groupnorm + gate (WMMA)
  {
    dim3 grid(N_ / 16 / 8, B_ * H_);
    retention_kernel<<<grid, 256, 0, stream>>>(qrot, krot, vT, gate, retg);
  }

  // 6) output projection (WMMA GEMM) -> fp32 out
  gemm_bf16_kernel<<<M_ / 16, 256, 0, stream>>>(retg, Wob, bo, out);
}
